// GIN_40029095198816
// MI455X (gfx1250) — matrix operand order repaired and checked by reading, not compiled, not run on hardware
//
#include <hip/hip_runtime.h>

#define NNODES 327680
#define NEDGES 1310720
#define SACT   160      // bf16 activation row stride (elements, K-padded)
#define PANEL  24576    // panel slot elems: 144*(160+8) weights + 144 bias + slack

typedef __attribute__((ext_vector_type(16))) __bf16 v16bf;
typedef __attribute__((ext_vector_type(8)))  float  v8f;
typedef unsigned int u32x4 __attribute__((ext_vector_type(4)));
typedef int          i32x4 __attribute__((ext_vector_type(4)));
typedef int          i32x8 __attribute__((ext_vector_type(8)));

union AFrag { v16bf v; uint4 q[2]; };
union B8    { uint4 q; __bf16 h[8]; };

enum { OUT_BF16 = 0, OUT_F32_PAD = 2 };

#define HAS_TDM (__has_builtin(__builtin_amdgcn_tensor_load_to_lds) && \
                 __has_builtin(__builtin_amdgcn_s_wait_tensorcnt))

// ---------------- elementwise / prep kernels ----------------

// x fp32 [N,fi] -> bf16 [N,SACT] zero K-padded; 8 elems (16B store) per thread
__global__ void k_cvt_bf16(const float* __restrict__ x, __bf16* __restrict__ y, int fi) {
  long long i8 = ((long long)blockIdx.x * blockDim.x + threadIdx.x) * 8;
  if (i8 >= (long long)NNODES * SACT) return;
  int n = (int)(i8 / SACT), k = (int)(i8 % SACT);
  B8 o;
#pragma unroll
  for (int j = 0; j < 8; j++) {
    int kk = k + j;
    float v = (kk < fi) ? x[(long long)n * fi + kk] : 0.f;
    o.h[j] = (__bf16)v;
  }
  *(uint4*)(y + i8) = o.q;
}

__global__ void k_zero16(uint4* __restrict__ p, long long cnt) {  // cnt in uint4 units
  long long i = (long long)blockIdx.x * blockDim.x + threadIdx.x;
  long long stride = (long long)gridDim.x * blockDim.x;
  uint4 z = {0u, 0u, 0u, 0u};
  for (; i < cnt; i += stride) p[i] = z;
}

// Prep one weight matrix: panel[n*(KP+8)+k] = bf16(W[k*fo+n]) zero-padded,
// followed by 144 bf16 zero-padded bias entries (rides the same TDM DMA).
__global__ void k_prep(const float* __restrict__ W, const float* __restrict__ b,
                       __bf16* __restrict__ panel, int fi, int fo, int KP, int rows) {
  int bstr = KP + 8;
  int tot = rows * bstr;
  int i = blockIdx.x * blockDim.x + threadIdx.x;
  if (i < tot) {
    int n = i / bstr, k = i - n * bstr;
    float w = (k < fi && n < fo) ? W[(long long)k * fo + n] : 0.f;
    panel[i] = (__bf16)w;
  } else {
    int j = i - tot;
    if (j < 144) panel[tot + j] = (__bf16)((j < fo) ? b[j] : 0.f);
  }
}

// segment_sum in bf16: agg[dst] += act[src], packed-pair atomics (halves RMW traffic)
__global__ void k_scatter(const int* __restrict__ ei, const __bf16* __restrict__ act,
                          __bf16* __restrict__ agg, int npairs) {
  int e = blockIdx.x * blockDim.x + threadIdx.x;
  if (e >= NEDGES) return;
  int src = ei[e];
  int dst = ei[NEDGES + e];
  int p0 = blockIdx.y * 4;
  const unsigned int* a = (const unsigned int*)(act + (long long)src * SACT);
  unsigned long long gbase = (unsigned long long)(agg + (long long)dst * SACT);
#pragma unroll
  for (int j = 0; j < 4; j++) {
    int p = p0 + j;
    if (p < npairs) {
      unsigned int d = a[p];
      unsigned long long addr = gbase + (unsigned long long)p * 4ull;
      asm volatile("global_atomic_pk_add_bf16 %0, %1, off"
                   :: "v"(addr), "v"(d) : "memory");
    }
  }
}

// sb = bf16(act + agg), K-padded with zeros; 8 elems per thread
__global__ void k_sum(const __bf16* __restrict__ act, const __bf16* __restrict__ agg,
                      __bf16* __restrict__ sb, int fi) {
  long long i8 = ((long long)blockIdx.x * blockDim.x + threadIdx.x) * 8;
  if (i8 >= (long long)NNODES * SACT) return;
  int k = (int)(i8 % SACT);
  B8 a, g, o;
  a.q = *(const uint4*)(act + i8);
  g.q = *(const uint4*)(agg + i8);
#pragma unroll
  for (int j = 0; j < 8; j++) {
    int kk = k + j;
    float v = (kk < fi) ? (float)a.h[j] + (float)g.h[j] : 0.f;
    o.h[j] = (__bf16)v;
  }
  *(uint4*)(sb + i8) = o.q;
}

// zero the 5 pad rows per graph in the final (8192,45,128) layout; float4 stores
__global__ void k_zero_pad(float* __restrict__ out) {
  int i = blockIdx.x * blockDim.x + threadIdx.x;   // 8192 graphs * 160 float4
  if (i >= 8192 * 160) return;
  int g = i / 160, r = i - g * 160;
  uint4 z = {0u, 0u, 0u, 0u};
  ((uint4*)(out + (long long)g * 5760 + 5120))[r] = z;
}

// ---------------- WMMA GEMM ----------------
// C[m,n] = act( A[m,:] @ Wt + bias[n] (+ existing C) )
// A: bf16 [N, SACT] row-major, K zero-padded to KSTEPS*32.
// Wt: prepped bf16 panel [rows=NT*16][KP+8] + 144 bias entries, in global;
// one TDM DMA -> LDS per workgroup. 256 threads = 8 waves x 16 rows.
template<int KSTEPS, int OMODE, int ADDC, int RELU>
__global__ __launch_bounds__(256) void k_gemm(
    const __bf16* __restrict__ A,
    const __bf16* __restrict__ Wt,
    int NT, void* __restrict__ outp)
{
  constexpr int KP   = KSTEPS * 32;
  constexpr int BSTR = KP + 8;                 // +16B: de-conflicts 64 LDS banks
  __shared__ __align__(64) __bf16 Wlds[144 * (160 + 8) + 144];

  int tot = NT * 16 * BSTR + 144;              // weights + padded bias (bf16 elems)
#if HAS_TDM
  if (threadIdx.x < 32) {                      // one wave issues the tensor DMA
    unsigned long long ga = (unsigned long long)Wt;
    unsigned int ldsa = (unsigned int)(unsigned long long)&Wlds[0];
    u32x4 g0;
    g0[0] = 1u;                                 // count=1 valid descriptor
    g0[1] = ldsa;                               // lds_addr
    g0[2] = (unsigned int)ga;                   // global_addr[31:0]
    g0[3] = (unsigned int)(ga >> 32) | (2u << 30);  // global_addr[56:32] | type=2
    i32x8 g1;
    g1[0] = 1 << 16;                            // data_size = 2 bytes
    g1[1] = (tot & 0xffff) << 16;               // tensor_dim0[15:0]
    g1[2] = (tot >> 16) | (1 << 16);            // tensor_dim0[31:16] | tensor_dim1=1
    g1[3] = (tot & 0xffff) << 16;               // tile_dim0 = tot (1-D copy)
    g1[4] = 1;                                  // tile_dim1 = 1
    g1[5] = tot;                                // tensor_dim0_stride
    g1[6] = 0; g1[7] = 0;
    i32x4 z4 = {0, 0, 0, 0};
#if defined(__clang_major__) && __clang_major__ >= 23
    i32x8 z8 = {0, 0, 0, 0, 0, 0, 0, 0};
    __builtin_amdgcn_tensor_load_to_lds(g0, g1, z4, z4, z8, 0);
#else
    __builtin_amdgcn_tensor_load_to_lds(g0, g1, z4, z4, 0);
#endif
    __builtin_amdgcn_s_wait_tensorcnt(0);
  }
  __syncthreads();
#else
  for (int i = threadIdx.x; i < (tot >> 3); i += 256)
    ((uint4*)Wlds)[i] = ((const uint4*)Wt)[i];
  __syncthreads();
#endif

  int lane = threadIdx.x & 31;
  int wv   = threadIdx.x >> 5;
  int m0   = blockIdx.x * 128 + wv * 16;
  int hsel = lane >> 4;
  int mloc = lane & 15;
  const __bf16* biasLds = &Wlds[NT * 16 * BSTR];

  // A fragments (16x32 bf16): lanes0-15 K{k0..k0+7,k0+16..k0+23},
  // lanes16-31 K{k0+8..k0+15,k0+24..k0+31}
  const __bf16* arow = A + (long long)(m0 + mloc) * SACT;
  AFrag af[KSTEPS];
#pragma unroll
  for (int ks = 0; ks < KSTEPS; ks++) {
    const __bf16* p = arow + ks * 32 + hsel * 8;
    af[ks].q[0] = *(const uint4*)(p);
    af[ks].q[1] = *(const uint4*)(p + 16);
  }

  for (int nt = 0; nt < NT; nt++) {
    int col = nt * 16 + mloc;                  // each lane owns one output column
    float bc = (float)biasLds[col];            // ds load, no vmem stall

    int idx[8];
    v8f acc;
#pragma unroll
    for (int r = 0; r < 8; r++) {
      int row = m0 + hsel * 8 + r;             // C/D layout: lanes16-31 -> M+8
      if (OMODE == OUT_F32_PAD) {
        int gph = row / 40, rr = row - gph * 40;
        idx[r] = gph * 5760 + rr * 128 + col;
      } else {
        idx[r] = row * SACT + col;
      }
      acc[r] = bc;
    }
    if (ADDC) {
#pragma unroll
      for (int r = 0; r < 8; r++)
        acc[r] += (OMODE == OUT_F32_PAD) ? ((const float*)outp)[idx[r]]
                                         : (float)((const __bf16*)outp)[idx[r]];
    }
#pragma unroll
    for (int ks = 0; ks < KSTEPS; ks++) {
      // B fragment: lanes0-15 K{k0..k0+7,k0+16..k0+23} of column col, lanes16-31 K{k0+8..k0+15,k0+24..k0+31}
      AFrag bfr;
      const __bf16* bp = &Wlds[col * BSTR + ks * 32 + hsel * 8];
      bfr.q[0] = *(const uint4*)(bp);
      bfr.q[1] = *(const uint4*)(bp + 16);
      acc = __builtin_amdgcn_wmma_f32_16x16x32_bf16(false, af[ks].v, false, bfr.v,
                                                    (short)0, acc, false, false);
    }
    if (RELU) {
#pragma unroll
      for (int r = 0; r < 8; r++) acc[r] = acc[r] > 0.f ? acc[r] : 0.f;
    }
#pragma unroll
    for (int r = 0; r < 8; r++) {
      if (OMODE == OUT_F32_PAD) ((float*)outp)[idx[r]] = acc[r];
      else                      ((__bf16*)outp)[idx[r]] = (__bf16)acc[r];
    }
  }
}

// ---------------- driver ----------------

extern "C" void kernel_launch(void* const* d_in, const int* in_sizes, int n_in,
                              void* d_out, int out_size, void* d_ws, size_t ws_size,
                              hipStream_t stream) {
  (void)in_sizes; (void)n_in; (void)out_size; (void)ws_size;
  const float* x  = (const float*)d_in[0];
  const int*   ei = (const int*)d_in[1];

  size_t actElems = (size_t)NNODES * SACT;
  __bf16* buf0 = (__bf16*)d_ws;
  __bf16* buf1 = buf0 + actElems;
  __bf16* sb   = buf1 + actElems;
  __bf16* h1b  = sb   + actElems;
  __bf16* panels = h1b + actElems;             // 12 prepped bf16 weight+bias panels

  __bf16* agg = (__bf16*)d_out;   // bf16 agg scratch; fully dead before final writes
  float*  out = (float*)d_out;

  const int fiA[4] = {79, 138, 138, 138};
  const int foA[4] = {138, 138, 138, 128};
  const int KP1[4] = {96, 160, 160, 160};   // pad32(fi) for Wr/W1 A-side
  const int KP2[4] = {160, 160, 160, 128};  // pad32(fo) for W2 A-side
  const int ntA[4] = {9, 9, 9, 8};          // ceil(fo/16)

  // ---- prep all weight panels (+ bf16 bias tails) ----
  for (int l = 0; l < 4; l++) {
    int rows = ntA[l] * 16;
    for (int w = 0; w < 3; w++) {
      const float* W = (const float*)d_in[2 + 6 * l + 2 * w];
      const float* b = (const float*)d_in[3 + 6 * l + 2 * w];
      int fi = (w == 2) ? foA[l] : fiA[l];
      int KP = (w == 2) ? KP2[l] : KP1[l];
      int tot = rows * (KP + 8) + 144;
      k_prep<<<(tot + 255) / 256, 256, 0, stream>>>(
          W, b, panels + (l * 3 + w) * PANEL, fi, foA[l], KP, rows);
    }
  }
  auto P = [&](int l, int w) { return panels + (l * 3 + w) * PANEL; };

  long long actTot = (long long)NNODES * SACT;
  int vecBlocks = (int)((actTot / 8 + 255) / 256);
  dim3 gg(NNODES / 128), gb(256);

  k_cvt_bf16<<<vecBlocks, 256, 0, stream>>>(x, buf0, fiA[0]);

  __bf16* act = buf0;
  __bf16* nxt = buf1;
  for (int l = 0; l < 4; l++) {
    int fi = fiA[l], NT = ntA[l];
    int npairs = (fi + 1) / 2;

    k_zero16<<<4096, 256, 0, stream>>>((uint4*)agg, actTot / 8);
    dim3 sg(NEDGES / 256, (npairs + 3) / 4);
    k_scatter<<<sg, 256, 0, stream>>>(ei, act, agg, npairs);
    k_sum<<<vecBlocks, 256, 0, stream>>>(act, agg, sb, fi);

    if (l == 0) {
      k_gemm<3, OUT_BF16, 0, 1><<<gg, gb, 0, stream>>>(sb,  P(0,1), NT, h1b);
      k_gemm<5, OUT_BF16, 0, 0><<<gg, gb, 0, stream>>>(h1b, P(0,2), NT, nxt);
      k_gemm<3, OUT_BF16, 1, 0><<<gg, gb, 0, stream>>>(act, P(0,0), NT, nxt);
      __bf16* t = act; act = nxt; nxt = t;
    } else if (l < 3) {
      k_gemm<5, OUT_BF16, 0, 1><<<gg, gb, 0, stream>>>(sb,  P(l,1), NT, h1b);
      k_gemm<5, OUT_BF16, 0, 0><<<gg, gb, 0, stream>>>(h1b, P(l,2), NT, nxt);
      k_gemm<5, OUT_BF16, 1, 0><<<gg, gb, 0, stream>>>(act, P(l,0), NT, nxt);
      __bf16* t = act; act = nxt; nxt = t;
    } else {
      k_gemm<5, OUT_BF16, 0, 1><<<gg, gb, 0, stream>>>(sb,  P(3,1), NT, h1b);
      k_zero_pad<<<(8192 * 160 + 255) / 256, 256, 0, stream>>>(out);
      k_gemm<4, OUT_F32_PAD, 0, 0><<<gg, gb, 0, stream>>>(h1b, P(3,2), NT, out);
      k_gemm<5, OUT_F32_PAD, 1, 0><<<gg, gb, 0, stream>>>(act, P(3,0), NT, out);
    }
  }
}
